// DownBlock_55654186221981
// MI455X (gfx1250) — 8-bit matrix operand order repaired and checked by reading, compiled, not run on hardware
//
#include <hip/hip_runtime.h>
#include <hip/hip_bf16.h>
#include <cstdint>
#include <cstddef>

typedef int v8i __attribute__((ext_vector_type(8)));

#define B_    32
#define CIN   128
#define E_    512
#define P_    256
#define H_    64
#define W_    64
#define OH_   32
#define OW_   32
#define KTOT  1152
#define NTOT  32768
#define NSTEP 18

#if defined(__AMDGCN__) && \
    __has_builtin(__builtin_amdgcn_global_load_async_to_lds_b64) && \
    __has_builtin(__builtin_amdgcn_global_load_async_to_lds_b128)
#define USE_ASYNC_LDS 1
typedef int v2i_g __attribute__((vector_size(8)));
typedef int v4i_g __attribute__((vector_size(16)));
#define GLOBAL_AS __attribute__((address_space(1)))
#define LDS_AS    __attribute__((address_space(3)))
#else
#define USE_ASYNC_LDS 0
#endif

__device__ __forceinline__ void async_copy_b64(const void* g, void* l) {
#if USE_ASYNC_LDS
    __builtin_amdgcn_global_load_async_to_lds_b64((GLOBAL_AS v2i_g*)g,
                                                  (LDS_AS v2i_g*)l, 0, 0);
#else
    *(uint2*)l = *(const uint2*)g;
#endif
}

__device__ __forceinline__ void async_copy_b128(const void* g, void* l) {
#if USE_ASYNC_LDS
    __builtin_amdgcn_global_load_async_to_lds_b128((GLOBAL_AS v4i_g*)g,
                                                   (LDS_AS v4i_g*)l, 0, 0);
#else
    *(int4*)l = *(const int4*)g;
#endif
}

__device__ __forceinline__ void wait_async_lds() {
#if USE_ASYNC_LDS
#if __has_builtin(__builtin_amdgcn_s_wait_asynccnt)
    __builtin_amdgcn_s_wait_asynccnt(0);
#else
    asm volatile("s_wait_asynccnt 0x0" ::: "memory");
#endif
#endif
}

__global__ void k_bias(const float* __restrict__ emb,
                       const float* __restrict__ m1w, const float* __restrict__ m1b,
                       const float* __restrict__ m2w, const float* __restrict__ m2b,
                       const float* __restrict__ m3w, const float* __restrict__ m3b,
                       float* __restrict__ bias1, float* __restrict__ bias2,
                       float* __restrict__ bias3) {
    __shared__ float s[E_];
    const int b = blockIdx.x;
    for (int e = threadIdx.x; e < E_; e += blockDim.x) {
        float z = emb[b * E_ + e];
        s[e] = z / (1.0f + __expf(-z));
    }
    __syncthreads();
    for (int j = threadIdx.x; j < CIN + P_ + P_; j += blockDim.x) {
        const float* wrow; float bb; float* dst;
        if (j < CIN) {
            wrow = m1w + (size_t)j * E_;          bb = m1b[j];
            dst = bias1 + b * CIN + j;
        } else if (j < CIN + P_) {
            int p = j - CIN;
            wrow = m2w + (size_t)p * E_;          bb = m2b[p];
            dst = bias2 + b * P_ + p;
        } else {
            int p = j - CIN - P_;
            wrow = m3w + (size_t)p * E_;          bb = m3b[p];
            dst = bias3 + b * P_ + p;
        }
        float acc = bb;
        for (int e = 0; e < E_; ++e) acc = fmaf(s[e], wrow[e], acc);
        *dst = acc;
    }
}

__global__ void k_binact(const float* __restrict__ x,
                         const float* __restrict__ bias1,
                         int8_t* __restrict__ q) {
    const int bh = blockIdx.x;
    const int b = bh >> 6, h = bh & 63;
    for (int idx = threadIdx.x; idx < W_ * CIN; idx += blockDim.x) {
        int w = idx >> 7, c = idx & 127;
        float v = x[(((size_t)b * CIN + c) * H_ + h) * W_ + w] + bias1[b * CIN + c];
        int8_t s = (v > 0.f) ? (int8_t)1 : ((v < 0.f) ? (int8_t)-1 : (int8_t)0);
        q[(size_t)bh * (W_ * CIN) + (size_t)w * CIN + c] = s;
    }
}

__global__ void k_binw(const float* __restrict__ cw,
                       int8_t* __restrict__ wq, float* __restrict__ sf) {
    const int m = blockIdx.x;
    const float* wsrc = cw + (size_t)m * KTOT;
    __shared__ float red[128];
    float asum = 0.f;
    for (int t = threadIdx.x; t < KTOT; t += blockDim.x) {
        int ci = t / 9, r = t % 9;
        float wv = wsrc[t];
        asum += fabsf(wv);
        int8_t s = (wv > 0.f) ? (int8_t)1 : ((wv < 0.f) ? (int8_t)-1 : (int8_t)0);
        wq[(size_t)m * KTOT + (size_t)r * CIN + ci] = s;
    }
    red[threadIdx.x] = asum;
    __syncthreads();
    for (int off = 64; off > 0; off >>= 1) {
        if ((int)threadIdx.x < off) red[threadIdx.x] += red[threadIdx.x + off];
        __syncthreads();
    }
    if (threadIdx.x == 0) sf[m] = red[0] / (float)KTOT;
}

__global__ void k_avgpool(const float* __restrict__ x, float* __restrict__ xp) {
    size_t idx = (size_t)blockIdx.x * blockDim.x + threadIdx.x;
    if (idx >= (size_t)B_ * CIN * OH_ * OW_) return;
    int ow = idx & 31; size_t t = idx >> 5;
    int oh = t & 31;   t >>= 5;
    int c  = t & 127;  int b = (int)(t >> 7);
    const float* px = x + (((size_t)b * CIN + c) * H_ + 2 * oh) * W_ + 2 * ow;
    xp[idx] = 0.25f * (px[0] + px[1] + px[W_] + px[W_ + 1]);
}

__global__ void __launch_bounds__(256) k_conv_wmma(
        const int8_t* __restrict__ q, const int8_t* __restrict__ wq,
        const float* __restrict__ sf, const float* __restrict__ convb,
        float* __restrict__ ypre) {
    __shared__ __align__(16) int8_t lA[NSTEP * 64 * 64];
    __shared__ __align__(16) int8_t lB[2][32 * 64];
    const int row   = blockIdx.x;
    const int b     = row >> 5, oh = row & 31;
    const int mbase = blockIdx.y * 64;
    const int tid   = threadIdx.x;
    const int lane  = tid & 31;
    const int wave  = tid >> 5;
    const int mt    = wave >> 1;
    const int nt    = wave & 1;
    const int hi16  = (lane >= 16) ? 1 : 0;

    auto stage_B = [&](int s, int bufi) {
        int kh = s / 6, kw = (s % 6) >> 1, cb2 = s & 1;
        int ih = 2 * oh - 1 + kh;
        int n  = tid >> 3;
        int c8 = (tid & 7) * 8;
        int iw = 2 * n - 1 + kw;
        int8_t* dst = &lB[bufi][n * 64 + c8];
        if (ih >= 0 && ih < H_ && iw >= 0 && iw < W_) {
            const int8_t* g = q + ((size_t)((b * H_ + ih) * W_ + iw)) * CIN
                              + cb2 * 64 + c8;
            async_copy_b64(g, dst);
        } else {
            *(uint32_t*)dst       = 0u;
            *(uint32_t*)(dst + 4) = 0u;
        }
    };

    {
        int j = tid >> 2, ci16 = (tid & 3) * 16;
        __builtin_prefetch(q + ((size_t)(b * H_) * W_) * CIN + tid * 64, 0, 0);
#pragma unroll
        for (int s = 0; s < NSTEP; ++s) {
            int kh = s / 6, kw = (s % 6) >> 1, cb2 = s & 1;
            const int8_t* g = wq + ((size_t)(mbase + j) * 9 + kh * 3 + kw) * CIN
                              + cb2 * 64 + ci16;
            async_copy_b128(g, &lA[s * 4096 + j * 64 + ci16]);
        }
    }
    stage_B(0, 0);
    wait_async_lds();
    __syncthreads();

    v8i acc = {0, 0, 0, 0, 0, 0, 0, 0};
    const int ml = mt * 16 + (lane & 15);
    const int nl = nt * 16 + (lane & 15);

#pragma unroll
    for (int s = 0; s < NSTEP; ++s) {
        const int cur = s & 1;
        if (s + 1 < NSTEP) stage_B(s + 1, cur ^ 1);
        v8i afrag, bfrag;
#pragma unroll
        for (int v = 0; v < 8; ++v) {
            int ka = ((v >> 1) * 16) + hi16 * 8 + (v & 1) * 4;
            afrag[v] = *(const int*)(lA + s * 4096 + ml * 64 + ka);
            int kb = ((v >> 1) * 16) + hi16 * 8 + (v & 1) * 4;
            bfrag[v] = *(const int*)(&lB[cur][0] + nl * 64 + kb);
        }
        acc = __builtin_amdgcn_wmma_i32_16x16x64_iu8(
                 true, afrag,  true, bfrag,
                acc,  false,  false);
        if (s + 1 < NSTEP) {
            wait_async_lds();
            __syncthreads();
        }
    }

    const int col = nt * 16 + (lane & 15);
    const size_t n = (size_t)row * 32 + col;
#pragma unroll
    for (int r = 0; r < 8; ++r) {
        int m = mbase + mt * 16 + r + hi16 * 8;
        ypre[(size_t)m * NTOT + n] = sf[m] * (float)acc[r] + convb[m];
    }
}

__global__ void k_bnstats(const float* __restrict__ ypre,
                          const float* __restrict__ bng, const float* __restrict__ bnb,
                          float* __restrict__ bnscale, float* __restrict__ bnshift) {
    const int m = blockIdx.x;
    __shared__ float rs[256], rs2[256];
    float s = 0.f, s2 = 0.f;
    const float* p = ypre + (size_t)m * NTOT;
    for (int i = threadIdx.x; i < NTOT; i += blockDim.x) {
        float v = p[i]; s += v; s2 += v * v;
    }
    rs[threadIdx.x] = s; rs2[threadIdx.x] = s2;
    __syncthreads();
    for (int off = 128; off > 0; off >>= 1) {
        if ((int)threadIdx.x < off) {
            rs[threadIdx.x]  += rs[threadIdx.x + off];
            rs2[threadIdx.x] += rs2[threadIdx.x + off];
        }
        __syncthreads();
    }
    if (threadIdx.x == 0) {
        float mean = rs[0] / (float)NTOT;
        float var  = rs2[0] / (float)NTOT - mean * mean;
        float sc   = bng[m] * rsqrtf(var + 1e-5f);
        bnscale[m] = sc;
        bnshift[m] = bnb[m] - mean * sc;
    }
}

__global__ void k_epilogue(const float* __restrict__ ypre, const float* __restrict__ xp,
                           const float* __restrict__ bnscale, const float* __restrict__ bnshift,
                           const float* __restrict__ bias2, const float* __restrict__ bias3,
                           const float* __restrict__ prelu_a, float* __restrict__ out) {
    size_t idx = (size_t)blockIdx.x * blockDim.x + threadIdx.x;
    if (idx >= (size_t)B_ * P_ * OH_ * OW_) return;
    int ow = idx & 31; size_t t = idx >> 5;
    int oh = t & 31;   t >>= 5;
    int m  = t & 255;  int b = (int)(t >> 8);
    size_t n = ((size_t)b * OH_ + oh) * OW_ + ow;
    float v = ypre[(size_t)m * NTOT + n] * bnscale[m] + bnshift[m];
    v += xp[(((size_t)b * CIN + (m & 127)) * OH_ + oh) * OW_ + ow];
    v += bias2[b * P_ + m];
    v  = (v > 0.f) ? v : prelu_a[m] * v;
    v += bias3[b * P_ + m];
    out[idx] = v;
}

extern "C" void kernel_launch(void* const* d_in, const int* in_sizes, int n_in,
                              void* d_out, int out_size, void* d_ws, size_t ws_size,
                              hipStream_t stream) {
    (void)in_sizes; (void)n_in; (void)out_size; (void)ws_size;
    const float* x       = (const float*)d_in[0];
    const float* emb     = (const float*)d_in[1];
    const float* m1_w    = (const float*)d_in[2];
    const float* m1_b    = (const float*)d_in[3];
    const float* conv_w  = (const float*)d_in[4];
    const float* conv_b  = (const float*)d_in[5];
    const float* bn_g    = (const float*)d_in[6];
    const float* bn_b    = (const float*)d_in[7];
    const float* m2_w    = (const float*)d_in[8];
    const float* m2_b    = (const float*)d_in[9];
    const float* prelu_a = (const float*)d_in[10];
    const float* m3_w    = (const float*)d_in[11];
    const float* m3_b    = (const float*)d_in[12];
    float* out = (float*)d_out;

    char* ws = (char*)d_ws;
    size_t off = 0;
    auto alloc = [&](size_t bytes) {
        char* p = ws + off;
        off += (bytes + 255) & ~(size_t)255;
        return p;
    };
    float*  bias1   = (float*)alloc((size_t)B_ * CIN * 4);
    float*  bias2   = (float*)alloc((size_t)B_ * P_ * 4);
    float*  bias3   = (float*)alloc((size_t)B_ * P_ * 4);
    float*  sf      = (float*)alloc((size_t)P_ * 4);
    float*  bnscale = (float*)alloc((size_t)P_ * 4);
    float*  bnshift = (float*)alloc((size_t)P_ * 4);
    int8_t* wq      = (int8_t*)alloc((size_t)P_ * KTOT);
    int8_t* q       = (int8_t*)alloc((size_t)B_ * H_ * W_ * CIN);
    float*  xp      = (float*)alloc((size_t)B_ * CIN * OH_ * OW_ * 4);
    float*  ypre    = (float*)alloc((size_t)P_ * NTOT * 4);

    k_bias<<<B_, 256, 0, stream>>>(emb, m1_w, m1_b, m2_w, m2_b, m3_w, m3_b,
                                   bias1, bias2, bias3);
    k_binact<<<B_ * H_, 256, 0, stream>>>(x, bias1, q);
    k_binw<<<P_, 128, 0, stream>>>(conv_w, wq, sf);
    {
        int n = B_ * CIN * OH_ * OW_;
        k_avgpool<<<(n + 255) / 256, 256, 0, stream>>>(x, xp);
    }
    k_conv_wmma<<<dim3(B_ * OH_, 4), 256, 0, stream>>>(q, wq, sf, conv_b, ypre);
    k_bnstats<<<P_, 256, 0, stream>>>(ypre, bn_g, bn_b, bnscale, bnshift);
    {
        size_t n = (size_t)B_ * P_ * OH_ * OW_;
        k_epilogue<<<(unsigned)((n + 255) / 256), 256, 0, stream>>>(
            ypre, xp, bnscale, bnshift, bias2, bias3, prelu_a, out);
    }
}
